// ScaledDotProductdilateAttention_58360015618326
// MI455X (gfx1250) — compile-verified, matrix operand order checked by reading, not run on hardware
//
#include <hip/hip_runtime.h>
#include <hip/hip_bf16.h>
#include <stdint.h>

typedef __attribute__((ext_vector_type(16))) __bf16 v16bf;
typedef __attribute__((ext_vector_type(8)))  __bf16 v8bf;
typedef __attribute__((ext_vector_type(8)))  float  v8f;
typedef __attribute__((ext_vector_type(4)))  float  v4f;
typedef __attribute__((ext_vector_type(4)))  int    v4i;

#define B_BATCH 2048
#define NTOK    49
#define DMODEL  512
#define NHEAD   8
#define DHEAD   64
#define MTOT    (B_BATCH * NTOK)   // 100352, divisible by 64

#define AS1 __attribute__((address_space(1)))
#define AS3 __attribute__((address_space(3)))

static __device__ __forceinline__ v8f zero8() {
  v8f z;
#pragma unroll
  for (int i = 0; i < 8; ++i) z[i] = 0.0f;
  return z;
}

// Assemble the 16-element bf16 WMMA operand from the two 16B chunks
// (ISA layout: v[0..7] = K chunk at 8*(lane/16), v[8..15] = same + 16).
static __device__ __forceinline__ v16bf join16(v8bf lo, v8bf hi) {
  v16bf r;
#pragma unroll
  for (int i = 0; i < 8; ++i) { r[i] = lo[i]; r[8 + i] = hi[i]; }
  return r;
}

// 16-byte async global->LDS copy (per-lane).  The builtin's signature is
// (global v4i* src, LDS v4i* dst, imm offset, imm cpol); integer round-trip
// casts produce the address-space-qualified pointers (LDS offset = low 32
// bits of the generic pointer per ISA 10.2).
static __device__ __forceinline__ void async_copy_b128(const __bf16* g, __bf16* l) {
#if __has_builtin(__builtin_amdgcn_global_load_async_to_lds_b128)
  __builtin_amdgcn_global_load_async_to_lds_b128(
      (AS1 v4i*)(uintptr_t)g, (AS3 v4i*)(unsigned)(uintptr_t)l, 0, 0);
#else
  asm volatile("global_load_async_to_lds_b128 %0, %1, off"
               :: "v"((unsigned)(uintptr_t)l),
                  "v"((unsigned long long)(uintptr_t)g)
               : "memory");
#endif
}

static __device__ __forceinline__ void wait_asynccnt0() {
#if __has_builtin(__builtin_amdgcn_s_wait_asynccnt)
  __builtin_amdgcn_s_wait_asynccnt(0);
#else
  asm volatile("s_wait_asynccnt 0x0" ::: "memory");
#endif
}

// ---------------------------------------------------------------------------
// f32 -> bf16 conversion (weights)
// ---------------------------------------------------------------------------
__global__ void cvt_f32_bf16_kernel(const float* __restrict__ s,
                                    __bf16* __restrict__ d, int n) {
  int i = blockIdx.x * 256 + threadIdx.x;
  if (i < n) d[i] = (__bf16)s[i];
}

// ---------------------------------------------------------------------------
// QKV projection: Y = X @ W^T + b.  mode(blockIdx.z): 0=Q, 1=K, 2=V.
// Block = 128 threads (4 waves); block tile 64(M) x 256(N); wave tile 16x256
// (16 accumulator tiles).  X is read only twice total (grid.y == 2); W strips
// are L2-resident (512 KB << 192 MB).
// Q,K stored (B,H,49,64) row-major bf16; V stored transposed (B,H,64d,64tok).
// ---------------------------------------------------------------------------
__global__ __launch_bounds__(128)
void qkv_proj_kernel(const float* __restrict__ Xq, const float* __restrict__ Xk,
                     const float* __restrict__ Xv,
                     const __bf16* __restrict__ Wqb, const __bf16* __restrict__ Wkb,
                     const __bf16* __restrict__ Wvb,
                     const float* __restrict__ bq, const float* __restrict__ bk,
                     const float* __restrict__ bv,
                     __bf16* __restrict__ Qo, __bf16* __restrict__ Ko,
                     __bf16* __restrict__ VTo) {
  const int mode = blockIdx.z;
  const float*  X    = (mode == 0) ? Xq : (mode == 1) ? Xk : Xv;
  const __bf16* W    = (mode == 0) ? Wqb : (mode == 1) ? Wkb : Wvb;
  const float*  bias = (mode == 0) ? bq : (mode == 1) ? bk : bv;

  const int wave = threadIdx.x >> 5;
  const int lane = threadIdx.x & 31;
  const int half = lane >> 4;
  const int l16  = lane & 15;

  const int m0 = blockIdx.x * 64 + wave * 16;
  const int n0 = blockIdx.y * 256;

  v8f acc[16];
#pragma unroll
  for (int t = 0; t < 16; ++t) acc[t] = zero8();

  const float* aptr = X + (size_t)(m0 + l16) * DMODEL + half * 8;

  for (int k0 = 0; k0 < DMODEL; k0 += 32) {
    if (k0 + 32 < DMODEL) __builtin_prefetch(aptr + k0 + 32, 0, 0);
    // A operand: 16 f32 -> bf16 in registers
    v4f a0 = *(const v4f*)(aptr + k0);
    v4f a1 = *(const v4f*)(aptr + k0 + 4);
    v4f a2 = *(const v4f*)(aptr + k0 + 16);
    v4f a3 = *(const v4f*)(aptr + k0 + 20);
    v16bf a;
#pragma unroll
    for (int i = 0; i < 4; ++i) {
      a[i]      = (__bf16)a0[i];
      a[4 + i]  = (__bf16)a1[i];
      a[8 + i]  = (__bf16)a2[i];
      a[12 + i] = (__bf16)a3[i];
    }
#pragma unroll
    for (int t = 0; t < 16; ++t) {
      const __bf16* wp = W + (size_t)(n0 + t * 16 + l16) * DMODEL + k0 + half * 8;
      v16bf bm = join16(*(const v8bf*)wp, *(const v8bf*)(wp + 16));
      acc[t] = __builtin_amdgcn_wmma_f32_16x16x32_bf16(
          false, a, false, bm, (short)0, acc[t], false, false);
    }
  }

  // Epilogue: bias add + scatter into per-head layout.
#pragma unroll
  for (int t = 0; t < 16; ++t) {
    const int col = n0 + t * 16 + l16;
    const int h  = col >> 6;
    const int dd = col & 63;
    const float bia = bias[col];
#pragma unroll
    for (int r = 0; r < 8; ++r) {
      const int mrow = m0 + r + half * 8;
      const int bb  = mrow / NTOK;
      const int tok = mrow - bb * NTOK;
      const float v = acc[t][r] + bia;
      if (mode == 2) {
        VTo[(((size_t)bb * NHEAD + h) * 64 + dd) * 64 + tok] = (__bf16)v;
      } else {
        __bf16* O = (mode == 0) ? Qo : Ko;
        O[(((size_t)bb * NHEAD + h) * NTOK + tok) * 64 + dd] = (__bf16)v;
      }
    }
  }
}

// ---------------------------------------------------------------------------
// Dilated attention per (b,h).  4 waves; wave w owns query rows [16w,16w+16).
// K and V^T tiles are staged to LDS with async global->LDS loads (shared by
// all waves); S = q k^T (WMMA), multiplicative dilate mask (zeros), scale 1/8,
// softmax over the 49 real keys (pad keys -inf), P in LDS bf16, O = P v.
// ---------------------------------------------------------------------------
__global__ __launch_bounds__(128)
void dilate_attn_kernel(const __bf16* __restrict__ Q, const __bf16* __restrict__ K,
                        const __bf16* __restrict__ VT, __bf16* __restrict__ A) {
  __shared__ __align__(16) __bf16 Ksh[64][72];
  __shared__ __align__(16) __bf16 Vsh[64][72];
  __shared__ __align__(16) __bf16 P[64][72];

  const int bh = blockIdx.x;
  const int b = bh >> 3;
  const int h = bh & 7;
  const int wave = threadIdx.x >> 5;
  const int lane = threadIdx.x & 31;
  const int half = lane >> 4;
  const int l16  = lane & 15;

  const __bf16* Qb = Q + (size_t)bh * NTOK * 64;
  const __bf16* Kb = K + (size_t)bh * NTOK * 64;
  const __bf16* Vb = VT + (size_t)bh * 64 * 64;

  // Stage K (64x64, rows >=49 are junk -> masked) and V^T (64x64) into LDS:
  // 512 16-byte chunks per tensor, 8 per thread, async.
  for (int i = threadIdx.x; i < 1024; i += 128) {
    const int tensor = i >> 9;          // 0 = K, 1 = V^T
    const int idx    = i & 511;
    const int row    = idx >> 3;
    const int c      = idx & 7;
    const __bf16* src = (tensor ? Vb : Kb) + (size_t)row * 64 + c * 8;
    __bf16* dst = tensor ? &Vsh[row][c * 8] : &Ksh[row][c * 8];
    async_copy_b128(src, dst);
  }
  wait_asynccnt0();
  __syncthreads();

  v8f s[4];
#pragma unroll
  for (int t = 0; t < 4; ++t) s[t] = zero8();

  const int arow = wave * 16 + l16;
#pragma unroll
  for (int k0 = 0; k0 < 64; k0 += 32) {
    const __bf16* qa = Qb + (size_t)arow * 64 + k0 + half * 8;
    v16bf a = join16(*(const v8bf*)qa, *(const v8bf*)(qa + 16));
#pragma unroll
    for (int t = 0; t < 4; ++t) {
      const __bf16* kp = &Ksh[t * 16 + l16][k0 + half * 8];
      v16bf bm = join16(*(const v8bf*)kp, *(const v8bf*)(kp + 16));
      s[t] = __builtin_amdgcn_wmma_f32_16x16x32_bf16(
          false, a, false, bm, (short)0, s[t], false, false);
    }
  }

  // Mask (multiplicative) + scale; pad columns -> -inf; row max.
  const float scale = 0.125f;  // 1/sqrt(64)
  float rmax[8], rsum[8];
#pragma unroll
  for (int r = 0; r < 8; ++r) rmax[r] = -__builtin_inff();

#pragma unroll
  for (int t = 0; t < 4; ++t) {
    const int kidx = t * 16 + l16;
    const int ki = kidx / 7, kj = kidx - ki * 7;
#pragma unroll
    for (int r = 0; r < 8; ++r) {
      const int qidx = wave * 16 + r + half * 8;
      const int qi = qidx / 7, qj = qidx - qi * 7;
      const int di = (qi > ki) ? qi - ki : ki - qi;
      const int dj = (qj > kj) ? qj - kj : kj - qj;
      const int cheb = (di > dj) ? di : dj;
      const int manh = di + dj;
      const bool R = (cheb == 1) && (manh == 1 || manh == 2);
      float v;
      if (kidx >= NTOK)      v = -__builtin_inff();
      else if (R)            v = s[t][r] * scale;
      else                   v = 0.0f;   // multiplicative mask: exp(0) joins softmax
      s[t][r] = v;
      rmax[r] = fmaxf(rmax[r], v);
    }
  }
#pragma unroll
  for (int r = 0; r < 8; ++r) {
#pragma unroll
    for (int off = 1; off < 16; off <<= 1)
      rmax[r] = fmaxf(rmax[r], __shfl_xor(rmax[r], off, 32));
    rsum[r] = 0.0f;
  }
#pragma unroll
  for (int t = 0; t < 4; ++t) {
    const int kidx = t * 16 + l16;
#pragma unroll
    for (int r = 0; r < 8; ++r) {
      const float p = (kidx < NTOK) ? __expf(s[t][r] - rmax[r]) : 0.0f;
      s[t][r] = p;
      rsum[r] += p;
    }
  }
#pragma unroll
  for (int r = 0; r < 8; ++r) {
#pragma unroll
    for (int off = 1; off < 16; off <<= 1)
      rsum[r] += __shfl_xor(rsum[r], off, 32);
    rsum[r] = 1.0f / rsum[r];
  }

  // Stage probabilities to LDS as bf16.
#pragma unroll
  for (int t = 0; t < 4; ++t)
#pragma unroll
    for (int r = 0; r < 8; ++r)
      P[wave * 16 + r + half * 8][t * 16 + l16] = (__bf16)(s[t][r] * rsum[r]);

  __syncthreads();

  v8f o[4];
#pragma unroll
  for (int t = 0; t < 4; ++t) o[t] = zero8();

#pragma unroll
  for (int k0 = 0; k0 < 64; k0 += 32) {
    const __bf16* pa = &P[wave * 16 + l16][k0 + half * 8];
    v16bf a = join16(*(const v8bf*)pa, *(const v8bf*)(pa + 16));
#pragma unroll
    for (int t = 0; t < 4; ++t) {
      const __bf16* vp = &Vsh[t * 16 + l16][k0 + half * 8];
      v16bf bm = join16(*(const v8bf*)vp, *(const v8bf*)(vp + 16));
      o[t] = __builtin_amdgcn_wmma_f32_16x16x32_bf16(
          false, a, false, bm, (short)0, o[t], false, false);
    }
  }

  // Store O into (B, tok, H*64) bf16.
#pragma unroll
  for (int t = 0; t < 4; ++t) {
    const int dd = t * 16 + l16;
#pragma unroll
    for (int r = 0; r < 8; ++r) {
      const int tok = wave * 16 + r + half * 8;
      if (tok < NTOK)
        A[((size_t)b * NTOK + tok) * (NHEAD * DHEAD) + h * 64 + dd] = (__bf16)o[t][r];
    }
  }
}

// ---------------------------------------------------------------------------
// Output projection: out = A(100352x512 bf16) @ Wo^T(64x512) + bo -> f32.
// ---------------------------------------------------------------------------
__global__ __launch_bounds__(128)
void out_proj_kernel(const __bf16* __restrict__ A, const __bf16* __restrict__ Wob,
                     const float* __restrict__ bo, float* __restrict__ out) {
  const int wave = threadIdx.x >> 5;
  const int lane = threadIdx.x & 31;
  const int half = lane >> 4;
  const int l16  = lane & 15;
  const int m0 = blockIdx.x * 64 + wave * 16;

  v8f acc[4];
#pragma unroll
  for (int t = 0; t < 4; ++t) acc[t] = zero8();

  const __bf16* ap = A + (size_t)(m0 + l16) * DMODEL + half * 8;
  for (int k0 = 0; k0 < DMODEL; k0 += 32) {
    if (k0 + 32 < DMODEL) __builtin_prefetch(ap + k0 + 32, 0, 0);
    v16bf a = join16(*(const v8bf*)(ap + k0), *(const v8bf*)(ap + k0 + 16));
#pragma unroll
    for (int t = 0; t < 4; ++t) {
      const __bf16* wp = Wob + (size_t)(t * 16 + l16) * DMODEL + k0 + half * 8;
      v16bf bm = join16(*(const v8bf*)wp, *(const v8bf*)(wp + 16));
      acc[t] = __builtin_amdgcn_wmma_f32_16x16x32_bf16(
          false, a, false, bm, (short)0, acc[t], false, false);
    }
  }
#pragma unroll
  for (int t = 0; t < 4; ++t) {
    const int col = t * 16 + l16;
    const float bia = bo[col];
#pragma unroll
    for (int r = 0; r < 8; ++r) {
      const int mrow = m0 + r + half * 8;
      out[(size_t)mrow * 64 + col] = acc[t][r] + bia;
    }
  }
}

// ---------------------------------------------------------------------------
extern "C" void kernel_launch(void* const* d_in, const int* in_sizes, int n_in,
                              void* d_out, int out_size, void* d_ws, size_t ws_size,
                              hipStream_t stream) {
  const float* q  = (const float*)d_in[0];
  const float* k  = (const float*)d_in[1];
  const float* v  = (const float*)d_in[2];
  const float* Wq = (const float*)d_in[3];
  const float* bq = (const float*)d_in[4];
  const float* Wk = (const float*)d_in[5];
  const float* bk = (const float*)d_in[6];
  const float* Wv = (const float*)d_in[7];
  const float* bv = (const float*)d_in[8];
  const float* Wo = (const float*)d_in[9];
  const float* bo = (const float*)d_in[10];

  char* ws = (char*)d_ws;
  const size_t szQKV = (size_t)B_BATCH * NHEAD * NTOK * 64 * 2;  // bf16
  const size_t szVT  = (size_t)B_BATCH * NHEAD * 64 * 64 * 2;
  const size_t szA   = (size_t)MTOT * DMODEL * 2;
  const size_t offQ  = 0;
  const size_t offK  = offQ + szQKV;
  const size_t offVT = offK + szQKV;
  const size_t offA  = offVT + szVT;
  const size_t offWq = offA + szA;
  const size_t offWk = offWq + (size_t)512 * 512 * 2;
  const size_t offWv = offWk + (size_t)512 * 512 * 2;
  const size_t offWo = offWv + (size_t)512 * 512 * 2;

  __bf16* Qb  = (__bf16*)(ws + offQ);
  __bf16* Kb  = (__bf16*)(ws + offK);
  __bf16* VTb = (__bf16*)(ws + offVT);
  __bf16* Ab  = (__bf16*)(ws + offA);
  __bf16* Wqb = (__bf16*)(ws + offWq);
  __bf16* Wkb = (__bf16*)(ws + offWk);
  __bf16* Wvb = (__bf16*)(ws + offWv);
  __bf16* Wob = (__bf16*)(ws + offWo);

  const int nw = 512 * 512;
  cvt_f32_bf16_kernel<<<(nw + 255) / 256, 256, 0, stream>>>(Wq, Wqb, nw);
  cvt_f32_bf16_kernel<<<(nw + 255) / 256, 256, 0, stream>>>(Wk, Wkb, nw);
  cvt_f32_bf16_kernel<<<(nw + 255) / 256, 256, 0, stream>>>(Wv, Wvb, nw);
  cvt_f32_bf16_kernel<<<(64 * 512 + 255) / 256, 256, 0, stream>>>(Wo, Wob, 64 * 512);

  qkv_proj_kernel<<<dim3(MTOT / 64, DMODEL / 256, 3), 128, 0, stream>>>(
      q, k, v, Wqb, Wkb, Wvb, bq, bk, bv, Qb, Kb, VTb);

  dilate_attn_kernel<<<B_BATCH * NHEAD, 128, 0, stream>>>(Qb, Kb, VTb, Ab);

  out_proj_kernel<<<MTOT / 64, 128, 0, stream>>>(Ab, Wob, bo, (float*)d_out);
}
